// NeRF_48447231098896
// MI455X (gfx1250) — matrix operand order repaired and checked by reading, not compiled, not run on hardware
//
#include <hip/hip_runtime.h>
#include <math.h>

// ---------------------------------------------------------------------------
// CDNA5 (gfx1250) fused NeRF inference: bf16 WMMA (v_wmma_f32_16x16x32_bf16),
// one ray per workgroup (8 wave32s x 16-row M tiles), activations in LDS.
// ---------------------------------------------------------------------------

typedef __bf16 v16bf __attribute__((ext_vector_type(16)));
typedef __bf16 v8bf  __attribute__((ext_vector_type(8)));
typedef float  v8f   __attribute__((ext_vector_type(8)));

#define PI_F 3.14159265358979323846f

// ---------------------------------------------------------------------------
// Weight prep: fp32 row-major [Ks x Ns] -> bf16 WMMA B-fragment layout,
// padded to [Kp x Np] (Kp multiple of 32, Np multiple of 16), zero-filled pad.
// Fragment f = ktile*ntiles + ntile occupies 512 bf16 (32x16 tile).
// Within a fragment, lane l (0..31) holds column n = (l&15) of two 8-long
// K strips at kbase and kbase+16, kbase = ktile*32 + (l>>4)*8  -> one 32B load per lane.
// ---------------------------------------------------------------------------
__global__ void nerf_prep_weights(const float* __restrict__ src,
                                  __bf16* __restrict__ dst,
                                  int Ks, int Ns, int Kp, int Np) {
  int i = blockIdx.x * blockDim.x + threadIdx.x;
  int total = Kp * Np;
  if (i >= total) return;
  int ntiles = Np >> 4;
  int f = i >> 9;            // 512 elems per fragment
  int r = i & 511;
  int lane = r >> 4;
  int e = r & 15;
  int ktile = f / ntiles;
  int ntile = f - ktile * ntiles;
  int n = ntile * 16 + (lane & 15);
  int k = ktile * 32 + (lane >> 4) * 8 + e + (e >= 8 ? 8 : 0);
  float v = (k < Ks && n < Ns) ? src[k * Ns + n] : 0.0f;
  dst[i] = (__bf16)v;
}

// Assemble an A-fragment (16x32 bf16, M x K) for a wave from row-major LDS.
// Per ISA 7.12.2: lane l (half h=l>>4, m=l&15) holds K strips
// [kcol0+8h, +8) and [kcol0+8h+16, +8)  -> two 16B LDS reads.
__device__ __forceinline__ v16bf make_a(const __bf16* p0, const __bf16* p1) {
  v8bf lo = *(const v8bf*)p0;
  v8bf hi = *(const v8bf*)p1;
  v16bf a;
#pragma unroll
  for (int i = 0; i < 8; ++i) { a[i] = lo[i]; a[i + 8] = hi[i]; }
  return a;
}

// One dense layer for this wave's 16 rows:
//   dst[16 x ntiles*16] = act( A[16 x ktiles*32] @ W + bias )
// A ktiles [0,split) come from a0 (stride a0s halves), rest from a1 (stride
// a1s; a1s==0 gives a row-broadcast input, used for the per-ray dir encoding).
template <bool RELU>
__device__ __forceinline__ void gemm_layer(
    const __bf16* a0, int a0s, int split,
    const __bf16* a1, int a1s,
    const __bf16* __restrict__ wf, const float* __restrict__ bias, int nbias,
    __bf16* dst, int dsts,
    int row0, int lane, int ktiles, int ntiles) {
  const int h  = lane >> 4;
  const int nl = lane & 15;
  const __bf16* ar0 = a0 + (row0 + nl) * a0s;
  const __bf16* ar1 = a1 + (row0 + nl) * a1s;
  for (int nt = 0; nt < ntiles; ++nt) {
    const int n = nt * 16 + nl;
    const float bv = bias[n < nbias ? n : 0];
    v8f acc;
#pragma unroll
    for (int i = 0; i < 8; ++i) acc[i] = bv;
    for (int kt = 0; kt < ktiles; ++kt) {
      const __bf16* s = (kt < split) ? (ar0 + kt * 32) : (ar1 + (kt - split) * 32);
      v16bf a = make_a(s + 8 * h, s + 8 * h + 16);
      v16bf b = *(const v16bf*)(wf + (size_t)(kt * ntiles + nt) * 512 + lane * 16);
      acc = __builtin_amdgcn_wmma_f32_16x16x32_bf16(false, a, false, b,
                                                    (short)0, acc, false, false);
    }
    // C/D layout: VGPR v -> row (v + 8h), column = lane&15 (ISA 7.12.2)
#pragma unroll
    for (int v = 0; v < 8; ++v) {
      float x = acc[v];
      if (RELU) x = x > 0.0f ? x : 0.0f;
      dst[(row0 + v + 8 * h) * dsts + nt * 16 + nl] = (__bf16)x;
    }
  }
}

// ---------------------------------------------------------------------------
// Fused kernel: block = 1 ray (128 samples), 256 threads = 8 wave32s.
// ---------------------------------------------------------------------------
__global__ __launch_bounds__(256)
void nerf_fused_kernel(
    const float* __restrict__ orig, const float* __restrict__ dirs,
    const float* __restrict__ tvals,
    const __bf16* __restrict__ W0p, const __bf16* __restrict__ Whp,
    const __bf16* __restrict__ W5p, const __bf16* __restrict__ W6p,
    const __bf16* __restrict__ W7p, const __bf16* __restrict__ Wsp,
    const __bf16* __restrict__ Wfp, const __bf16* __restrict__ Wmp,
    const __bf16* __restrict__ Wrp,
    const float* __restrict__ b0, const float* __restrict__ bh,
    const float* __restrict__ b5, const float* __restrict__ b6,
    const float* __restrict__ b7, const float* __restrict__ bs,
    const float* __restrict__ bfe, const float* __restrict__ bm,
    const float* __restrict__ br, float* __restrict__ out) {
  extern __shared__ char smem[];
  __bf16* PX   = (__bf16*)smem;          // [128][64]  pos-enc (pad 60->64)
  __bf16* HA   = PX + 128 * 64;          // [128][256] activations ping
  __bf16* HB   = HA + 128 * 256;         // [128][256] activations pong
  __bf16* PDsh = HB + 128 * 256;         // [32]       dir-enc (per-ray, bcast)
  float*  SIG  = (float*)(PDsh + 32);    // [128]
  float*  RGBs = SIG + 128;              // [128][4]
  float*  TVAL = RGBs + 128 * 4;         // [128]

  const int ray = blockIdx.x;
  const int tid = threadIdx.x;

  // -------- Phase 0: sample points + positional encoding --------
  {
    const int r = tid & 127, part = tid >> 7;
    const float tv = tvals[ray * 128 + r];
    const float ox = orig[ray * 3 + 0], oy = orig[ray * 3 + 1], oz = orig[ray * 3 + 2];
    const float dx = dirs[ray * 3 + 0], dy = dirs[ray * 3 + 1], dz = dirs[ray * 3 + 2];
    const float xx = ox + tv * dx, xy = oy + tv * dy, xz = oz + tv * dz;
    if (part == 0) {
      TVAL[r] = tv;
#pragma unroll
      for (int j = 0; j < 4; ++j) PX[r * 64 + 60 + j] = (__bf16)0.0f;  // K pad
    }
#pragma unroll
    for (int ii = 0; ii < 5; ++ii) {
      const int i = part * 5 + ii;                 // freq index 0..9
      const float f = (float)(1 << i) * PI_F;
      __bf16* row = PX + r * 64 + i * 6;
      row[0] = (__bf16)__sinf(f * xx);
      row[1] = (__bf16)__sinf(f * xy);
      row[2] = (__bf16)__sinf(f * xz);
      row[3] = (__bf16)__cosf(f * xx);
      row[4] = (__bf16)__cosf(f * xy);
      row[5] = (__bf16)__cosf(f * xz);
    }
    if (part == 1 && r < 32) {                     // per-ray dir encoding (24+pad)
      float val = 0.0f;
      if (r < 24) {
        const int i = r / 6, c = r % 6;
        const float f = (float)(1 << i) * PI_F;
        const float comp = ((c % 3) == 0) ? dx : (((c % 3) == 1) ? dy : dz);
        val = (c < 3) ? __sinf(f * comp) : __cosf(f * comp);
      }
      PDsh[r] = (__bf16)val;
    }
    __builtin_prefetch(W0p + (tid << 5), 0, 0);    // warm weights into caches
    __builtin_prefetch(Whp + (tid << 5), 0, 0);
  }
  __syncthreads();

  // -------- Phase 1: MLP trunk (no barriers: each wave owns its 16 rows) ----
  const int lane = tid & 31;
  const int row0 = (tid >> 5) * 16;
  const int h  = lane >> 4;
  const int nl = lane & 15;

  gemm_layer<true >(PX, 64, 2,  PX, 0,  W0p,            b0,       256, HA, 256, row0, lane, 2,  16);
  gemm_layer<true >(HA, 256, 8, HA, 0,  Whp + 0 * 65536, bh + 0,   256, HB, 256, row0, lane, 8,  16);
  gemm_layer<true >(HB, 256, 8, HB, 0,  Whp + 1 * 65536, bh + 256, 256, HA, 256, row0, lane, 8,  16);
  gemm_layer<true >(HA, 256, 8, HA, 0,  Whp + 2 * 65536, bh + 512, 256, HB, 256, row0, lane, 8,  16);
  gemm_layer<true >(HB, 256, 8, HB, 0,  Whp + 3 * 65536, bh + 768, 256, HA, 256, row0, lane, 8,  16);
  gemm_layer<true >(HA, 256, 8, PX, 64, W5p,            b5,       256, HB, 256, row0, lane, 10, 16); // skip: [h|px]
  gemm_layer<true >(HB, 256, 8, HB, 0,  W6p,            b6,       256, HA, 256, row0, lane, 8,  16);
  gemm_layer<true >(HA, 256, 8, HA, 0,  W7p,            b7,       256, HB, 256, row0, lane, 8,  16);

  // sigma head: 256 -> 1 (N padded to 16, only n==0 valid)
  {
    const float bv = bs[0];
    v8f acc;
#pragma unroll
    for (int i = 0; i < 8; ++i) acc[i] = bv;
    const __bf16* s = HB + (row0 + nl) * 256;
    for (int kt = 0; kt < 8; ++kt) {
      v16bf a = make_a(s + kt * 32 + 8 * h, s + kt * 32 + 8 * h + 16);
      v16bf b = *(const v16bf*)(Wsp + (size_t)kt * 512 + lane * 16);
      acc = __builtin_amdgcn_wmma_f32_16x16x32_bf16(false, a, false, b,
                                                    (short)0, acc, false, false);
    }
    if (nl == 0) {
#pragma unroll
      for (int v = 0; v < 8; ++v) {
        float x = acc[v];
        SIG[row0 + v + 8 * h] = x > 0.0f ? x : 0.0f;   // relu(density)
      }
    }
  }

  // feature head (no relu) then color MLP with dir-enc concat (broadcast rows)
  gemm_layer<false>(HB, 256, 8, HB, 0,   Wfp, bfe, 256, HA, 256, row0, lane, 8, 16);
  gemm_layer<true >(HA, 256, 8, PDsh, 0, Wmp, bm,  128, HB, 256, row0, lane, 9, 8); // 288->128, out in cols 0..127

  // rgb head: 128 -> 3 (N padded to 16), sigmoid
  {
    const float bv = (nl < 3) ? br[nl] : 0.0f;
    v8f acc;
#pragma unroll
    for (int i = 0; i < 8; ++i) acc[i] = bv;
    const __bf16* s = HB + (row0 + nl) * 256;
    for (int kt = 0; kt < 4; ++kt) {
      v16bf a = make_a(s + kt * 32 + 8 * h, s + kt * 32 + 8 * h + 16);
      v16bf b = *(const v16bf*)(Wrp + (size_t)kt * 512 + lane * 16);
      acc = __builtin_amdgcn_wmma_f32_16x16x32_bf16(false, a, false, b,
                                                    (short)0, acc, false, false);
    }
    if (nl < 3) {
#pragma unroll
      for (int v = 0; v < 8; ++v)
        RGBs[(row0 + v + 8 * h) * 4 + nl] = 1.0f / (1.0f + __expf(-acc[v]));
    }
  }

  __syncthreads();

  // -------- Phase 2: volume rendering along the ray (one thread / channel) --
  if (tid < 3) {
    const int c = tid;
    float Tacc = 1.0f, Cc = 0.0f, wsum = 0.0f;
    for (int r = 0; r < 128; ++r) {
      const float delta = (r < 127) ? (TVAL[r + 1] - TVAL[r]) : 1e8f;
      const float e = __expf(-SIG[r] * delta);
      const float wgt = Tacc * (1.0f - e);
      Cc += wgt * RGBs[r * 4 + c];
      wsum += wgt;
      Tacc *= e;
    }
    out[ray * 3 + c] = Cc + (1.0f - wsum) * 1.0f;      // white background
  }
}

// ---------------------------------------------------------------------------
extern "C" void kernel_launch(void* const* d_in, const int* in_sizes, int n_in,
                              void* d_out, int out_size, void* d_ws, size_t ws_size,
                              hipStream_t stream) {
  const float* o  = (const float*)d_in[0];
  const float* d  = (const float*)d_in[1];
  const float* t  = (const float*)d_in[2];
  const float* W0 = (const float*)d_in[3];
  const float* b0 = (const float*)d_in[4];
  const float* Wh = (const float*)d_in[5];
  const float* bh = (const float*)d_in[6];
  const float* W5 = (const float*)d_in[7];
  const float* b5 = (const float*)d_in[8];
  const float* W6 = (const float*)d_in[9];
  const float* b6 = (const float*)d_in[10];
  const float* W7 = (const float*)d_in[11];
  const float* b7 = (const float*)d_in[12];
  const float* Ws = (const float*)d_in[13];
  const float* bs = (const float*)d_in[14];
  const float* Wf = (const float*)d_in[15];
  const float* bf = (const float*)d_in[16];
  const float* Wm = (const float*)d_in[17];
  const float* bm = (const float*)d_in[18];
  const float* Wr = (const float*)d_in[19];
  const float* br = (const float*)d_in[20];
  float* out = (float*)d_out;

  const int B = in_sizes[0] / 3;  // 2048 rays

  // Workspace carve-up: bf16 fragment-layout weights (padded shapes).
  __bf16* ws  = (__bf16*)d_ws;
  __bf16* W0p = ws;                      // 64  x 256
  __bf16* Whp = W0p + 64 * 256;          // 4 x (256 x 256)
  __bf16* W5p = Whp + 4 * 256 * 256;     // 320 x 256
  __bf16* W6p = W5p + 320 * 256;         // 256 x 256
  __bf16* W7p = W6p + 256 * 256;         // 256 x 256
  __bf16* Wsp = W7p + 256 * 256;         // 256 x 16
  __bf16* Wfp = Wsp + 256 * 16;          // 256 x 256
  __bf16* Wmp = Wfp + 256 * 256;         // 288 x 128
  __bf16* Wrp = Wmp + 288 * 128;         // 128 x 16

  auto prep = [&](const float* s, __bf16* dp, int Ks, int Ns, int Kp, int Np) {
    int total = Kp * Np;
    nerf_prep_weights<<<(total + 255) / 256, 256, 0, stream>>>(s, dp, Ks, Ns, Kp, Np);
  };
  prep(W0, W0p, 60, 256, 64, 256);
  for (int i = 0; i < 4; ++i)
    prep(Wh + i * 256 * 256, Whp + i * 256 * 256, 256, 256, 256, 256);
  prep(W5, W5p, 316, 256, 320, 256);
  prep(W6, W6p, 256, 256, 256, 256);
  prep(W7, W7p, 256, 256, 256, 256);
  prep(Ws, Wsp, 256, 1, 256, 16);
  prep(Wf, Wfp, 256, 256, 256, 256);
  prep(Wm, Wmp, 280, 128, 288, 128);
  prep(Wr, Wrp, 128, 3, 128, 16);

  // Dynamic LDS: PX + HA + HB + PD (bf16) + SIG + RGB + TVAL (f32) = 150592 B
  const size_t smem = (size_t)(128 * 64 + 128 * 256 + 128 * 256 + 32) * sizeof(__bf16)
                    + (size_t)(128 + 128 * 4 + 128) * sizeof(float);
  nerf_fused_kernel<<<B, 256, smem, stream>>>(
      o, d, t, W0p, Whp, W5p, W6p, W7p, Wsp, Wfp, Wmp, Wrp,
      b0, bh, b5, b6, b7, bs, bf, bm, br, out);
}
